// SelfNormAttention_25537875542112
// MI455X (gfx1250) — matrix operand order repaired and checked by reading, not compiled, not run on hardware
//
#include <hip/hip_runtime.h>
#include <math.h>

// Problem constants (from reference): B=4, S=2048, H=8, D=64
#define BB 4
#define SS 2048
#define HH 8
#define DD 64
#define CC 64            // chunk length
#define NC (SS / CC)     // 32 chunks
#define LDP 72           // LDS row stride in bf16 elements (144B, 16B aligned, bank-skewed)

typedef unsigned short u16;
typedef __attribute__((ext_vector_type(8)))  unsigned short u16x8;
typedef __attribute__((ext_vector_type(16))) __bf16         v16bf;
typedef __attribute__((ext_vector_type(8)))  float          v8f;

union FragU {
    struct { u16x8 lo; u16x8 hi; } s;
    v16bf v;
};

// float -> bf16 round-to-nearest-even (bit trick; NaN edge cases ignored)
__device__ __forceinline__ u16 f2bf(float f) {
    unsigned u = __float_as_uint(f);
    unsigned r = u + 0x7FFFu + ((u >> 16) & 1u);
    return (u16)(r >> 16);
}

// elu(x)+1  ==  x>0 ? x+1 : exp(x)
__device__ __forceinline__ float elup1(float x) {
    return x > 0.0f ? x + 1.0f : __expf(x);
}

// ---- WMMA fragment loaders (16-bit operands, 16x16x32, wave32) ----
// A matrix 16x32 (MxK), LDS stored row-major [m][k]:
//   lane<16 : row M=lane,    elems 0..7 = K=ks*32+0..7,   elems 8..15 = K=ks*32+16..23
//   lane>=16: row M=lane-16, elems 0..7 = K=ks*32+8..15,  elems 8..15 = K=ks*32+24..31
__device__ __forceinline__ v16bf load_a_frag(const u16* lds, int mrow, int ks, int lane) {
    const int hi = (lane >> 4) & 1;
    const u16* p = lds + (mrow + (lane & 15)) * LDP + ks * 32 + hi * 8;
    FragU f;
    f.s.lo = *(const u16x8*)(p);
    f.s.hi = *(const u16x8*)(p + 16);
    return f.v;
}

// B matrix 32x16 (KxN), LDS stored TRANSPOSED row-major [n][k]:
//   lane<16 : col N=lane,    elems 0..7 = K=ks*32+0..7,   elems 8..15 = K=ks*32+16..23
//   lane>=16: col N=lane-16, elems 0..7 = K=ks*32+8..15,  elems 8..15 = K=ks*32+24..31
__device__ __forceinline__ v16bf load_b_frag(const u16* ldsT, int ncol, int ks, int lane) {
    const int hi = (lane >> 4) & 1;
    const u16* p = ldsT + (ncol + (lane & 15)) * LDP + ks * 32 + hi * 8;
    FragU f;
    f.s.lo = *(const u16x8*)(p);
    f.s.hi = *(const u16x8*)(p + 16);
    return f.v;
}

__device__ __forceinline__ v8f wmma_bf16(v16bf a, v16bf b, v8f c) {
    return __builtin_amdgcn_wmma_f32_16x16x32_bf16(
        /*neg_a=*/false, a, /*neg_b=*/false, b,
        /*c_mod=*/(short)0, c, /*reuse_a=*/false, /*reuse_b=*/false);
}

// =====================================================================
// Kernel 1: per-chunk state  S_c = K_c^T @ V_c   (D x D, f32) -> ws
// ws layout: [b][h][c][d][t]  (4096 f32 per (b,h,c))
// =====================================================================
__global__ void __launch_bounds__(256)
k1_chunk_kv(const float* __restrict__ qk, const float* __restrict__ v,
            float* __restrict__ ws) {
    const int c = blockIdx.x, h = blockIdx.y, b = blockIdx.z;
    const int s0 = c * CC;
    const int tid = threadIdx.x;

    __shared__ __align__(16) u16 ktr[DD * LDP]; // ktr[d][s] = bf16(elup1(k[s][d]) * scale)
    __shared__ __align__(16) u16 vtr[DD * LDP]; // vtr[t][s] = bf16(v[s][t])

    {
        const int i = tid >> 2;          // local sequence position 0..63
        const int g = tid & 3;           // 16-element column group
        const float* krow = qk + ((((size_t)b * SS + (s0 + i)) * 2 + 1) * HH + h) * DD;
        const float* vrow = v  + (((size_t)b * SS + (s0 + i)) * HH + h) * DD;
#pragma unroll
        for (int j = 0; j < 16; ++j) {
            const int d = g * 16 + j;
            ktr[d * LDP + i] = f2bf(elup1(krow[d]) * 0.125f); // 1/sqrt(64)
            vtr[d * LDP + i] = f2bf(vrow[d]);
        }
    }
    __syncthreads();

    const int wave = tid >> 5;
    const int lane = tid & 31;
    const int hi   = (lane >> 4) & 1;

    float* dst = ws + ((((size_t)b * HH + h) * NC) + c) * (DD * DD);

#pragma unroll
    for (int t = 0; t < 2; ++t) {
        const int tile = wave * 2 + t;     // 0..15
        const int mi = tile >> 2;          // d-tile
        const int ni = tile & 3;           // t-tile
        v8f acc = {};
#pragma unroll
        for (int ks = 0; ks < 2; ++ks) {   // K dimension = 64 sequence positions
            v16bf a = load_a_frag(ktr, mi * 16, ks, lane);  // A[m=d][k=s]
            v16bf bf = load_b_frag(vtr, ni * 16, ks, lane); // B[k=s][n=t]
            acc = wmma_bf16(a, bf, acc);
        }
#pragma unroll
        for (int r = 0; r < 8; ++r) {
            const int rl = r + hi * 8;
            dst[(mi * 16 + rl) * DD + ni * 16 + (lane & 15)] = acc[r];
        }
    }
}

// =====================================================================
// Kernel 2: in-place inclusive prefix sum of chunk states over c
// =====================================================================
__global__ void __launch_bounds__(256)
k2_prefix(float* __restrict__ ws) {
    const int bh = blockIdx.x;           // 0..31
    const int tid = threadIdx.x;         // each thread owns 16 of 4096 elements
    float* base = ws + (size_t)bh * NC * (DD * DD) + tid * 16;
    float acc[16];
#pragma unroll
    for (int j = 0; j < 16; ++j) acc[j] = 0.0f;
    for (int c = 0; c < NC; ++c) {
        float* p = base + (size_t)c * (DD * DD);
#pragma unroll
        for (int j = 0; j < 16; ++j) {
            acc[j] += p[j];
            p[j] = acc[j];
        }
    }
}

// =====================================================================
// Kernel 3: out_chunk = norm * ( Q @ State_{c-1} + causal(Q K^T) @ V )
// =====================================================================
__global__ void __launch_bounds__(256)
k3_out(const float* __restrict__ qk, const float* __restrict__ v,
       const float* __restrict__ nvec, const float* __restrict__ offset,
       const float* __restrict__ ws, float* __restrict__ out) {
    const int c = blockIdx.x, h = blockIdx.y, b = blockIdx.z;
    const int s0 = c * CC;
    const int tid = threadIdx.x;

    __shared__ __align__(16) u16 qs[DD * LDP];  // q[i][d]          (A side)
    __shared__ __align__(16) u16 ksm[DD * LDP]; // k[j][d]*scale    (== B^T for Q K^T)
    __shared__ __align__(16) u16 vt[DD * LDP];  // v^T: vt[t][s]    (B side for P@V)
    __shared__ __align__(16) u16 st[DD * LDP];  // state^T: st[t][d](B side for Q@S)
    __shared__ __align__(16) u16 ps[DD * LDP];  // P[i][j] masked, bf16 (A side)
    __shared__ float nrm[CC];

    {
        const int i = tid >> 2;
        const int g = tid & 3;
        const float* qrow = qk + ((((size_t)b * SS + (s0 + i)) * 2 + 0) * HH + h) * DD;
        const float* krow = qrow + (size_t)HH * DD;
        const float* vrow = v + (((size_t)b * SS + (s0 + i)) * HH + h) * DD;
        const float* sp = ws + ((((size_t)b * HH + h) * NC) + (c > 0 ? c - 1 : 0)) * (DD * DD);
        const bool has_state = (c > 0);
#pragma unroll
        for (int j = 0; j < 16; ++j) {
            const int d = g * 16 + j;
            qs [i * LDP + d] = f2bf(elup1(qrow[d]));
            ksm[i * LDP + d] = f2bf(elup1(krow[d]) * 0.125f);
            vt [d * LDP + i] = f2bf(vrow[d]);
            // here i plays the role of feature index "din", d the output index "t"
            st [d * LDP + i] = has_state ? f2bf(sp[i * DD + d]) : (u16)0;
        }
        if (tid < CC) {
            const float y = nvec[((size_t)b * SS + (s0 + tid)) * HH + h] + offset[h];
            const float spl = fmaxf(y, 0.0f) + log1pf(__expf(-fabsf(y))); // softplus, stable
            nrm[tid] = __expf(-spl);
        }
    }
    __syncthreads();

    const int wave = tid >> 5;
    const int lane = tid & 31;
    const int hi   = (lane >> 4) & 1;
    const int col  = lane & 15;

    // ---- Phase A: P = causal(Q K^T), stored to LDS as bf16 ----
#pragma unroll
    for (int t = 0; t < 2; ++t) {
        const int tile = wave * 2 + t;
        const int mi = tile >> 2;
        const int nj = tile & 3;
        v8f acc = {};
        if (nj <= mi) {
#pragma unroll
            for (int ks = 0; ks < 2; ++ks) {
                v16bf a  = load_a_frag(qs, mi * 16, ks, lane);   // Q rows
                v16bf bf = load_b_frag(ksm, nj * 16, ks, lane);  // K rows == B^T
                acc = wmma_bf16(a, bf, acc);
            }
        }
#pragma unroll
        for (int r = 0; r < 8; ++r) {
            const int rl = r + hi * 8;
            float pv = acc[r];
            if (nj == mi && col > rl) pv = 0.0f;  // causal mask on diagonal tile
            ps[(mi * 16 + rl) * LDP + nj * 16 + col] = f2bf(pv);
        }
    }
    __syncthreads();

    // ---- Phase B: out = Q @ State + P @ V, scaled by norm ----
#pragma unroll
    for (int t = 0; t < 2; ++t) {
        const int tile = wave * 2 + t;
        const int mi = tile >> 2;
        const int ni = tile & 3;
        v8f acc = {};
#pragma unroll
        for (int ks = 0; ks < 2; ++ks) {       // inter-chunk: Q @ State
            v16bf a  = load_a_frag(qs, mi * 16, ks, lane);
            v16bf bf = load_b_frag(st, ni * 16, ks, lane);
            acc = wmma_bf16(a, bf, acc);
        }
#pragma unroll
        for (int ks = 0; ks < 2; ++ks) {       // intra-chunk: P @ V
            v16bf a  = load_a_frag(ps, mi * 16, ks, lane);
            v16bf bf = load_b_frag(vt, ni * 16, ks, lane);
            acc = wmma_bf16(a, bf, acc);
        }
#pragma unroll
        for (int r = 0; r < 8; ++r) {
            const int rl  = r + hi * 8;
            const int row = mi * 16 + rl;                 // local sequence pos
            const int d   = ni * 16 + col;                // output feature
            out[(((size_t)b * SS + (s0 + row)) * HH + h) * DD + d] = nrm[row] * acc[r];
        }
    }
}

extern "C" void kernel_launch(void* const* d_in, const int* in_sizes, int n_in,
                              void* d_out, int out_size, void* d_ws, size_t ws_size,
                              hipStream_t stream) {
    const float* qk     = (const float*)d_in[0];  // (B,S,2,H,D)
    const float* v      = (const float*)d_in[1];  // (B,S,H,D)
    const float* nvec   = (const float*)d_in[2];  // (B,S,H)
    const float* offset = (const float*)d_in[3];  // (H)
    float* out = (float*)d_out;                   // (B,S,H,D)
    float* ws  = (float*)d_ws;                    // needs B*H*NC*D*D*4 = 16 MiB

    dim3 grid(NC, HH, BB);   // (32, 8, 4) = 1024 blocks
    dim3 block(256);         // 8 wave32 waves

    k1_chunk_kv<<<grid, block, 0, stream>>>(qk, v, ws);
    k2_prefix<<<dim3(BB * HH), block, 0, stream>>>(ws);
    k3_out<<<grid, block, 0, stream>>>(qk, v, nvec, offset, ws, out);
}
